// NER_LSTM_CRF_62045097558493
// MI455X (gfx1250) — matrix operand order repaired and checked by reading, not compiled, not run on hardware
//
#include <hip/hip_runtime.h>
#include <hip/hip_bf16.h>

// Problem constants (from reference): V=50000, E=128, H=256, HD=128, K=9, B=128, T=512
#define VSZ    50000
#define EDIM   128
#define HDIM   256
#define HD     128
#define KTAGS  9
#define BB     128
#define TT     512
#define G4     512   // 4*HD gate width

typedef __attribute__((ext_vector_type(16))) __bf16 v16bf;
typedef __attribute__((ext_vector_type(16))) short  v16s;
typedef __attribute__((ext_vector_type(8)))  float  v8f;
typedef __attribute__((ext_vector_type(4)))  unsigned int v4u;
typedef __attribute__((ext_vector_type(8)))  int    v8i;
typedef __attribute__((ext_vector_type(4)))  int    v4i;

// LDS layout (byte offsets inside dynamic shared). 310 KB <= 320 KB/WGP.
#define OFF_WIH   0         // [512][128] bf16  (128 KB)
#define OFF_WHH   131072    // [512][128] bf16  (128 KB)
#define OFF_E0    262144    // [16][128] bf16 tile, buffer 0 (4 KB)
#define OFF_E1    266240    // [16][128] bf16 tile, buffer 1 (4 KB)
#define OFF_H     270336    // [16][128] bf16   (4 KB)
#define OFF_C     274432    // [16][128] f32    (8 KB)
#define OFF_G     282624    // [16][512] f32 gates (32 KB)
#define OFF_BIAS  315392    // [512] f32        (2 KB)
#define SMEM_BYTES 317440

#if __has_builtin(__builtin_amdgcn_tensor_load_to_lds)
#define USE_TDM 1
#else
#define USE_TDM 0
#endif

__device__ __forceinline__ unsigned short f32_to_bf16(float f) {
  unsigned int u = __float_as_uint(f);
  u += 0x7FFFu + ((u >> 16) & 1u);           // round-to-nearest-even
  return (unsigned short)(u >> 16);
}

#if USE_TDM
// ---------------------------------------------------------------------------
// TDM gather-mode load: 16 rows (16-bit row indices = token ids) of 128 bf16
// elements from the bf16 embedding table -> LDS[lds_off]. Async (TENSORcnt).
// Descriptor packing per CDNA5 ISA "Tensor DMA Descriptor" Groups 0-3.
// This toolchain's builtin takes 6 args: (g0, g1, g2, g3, <unused v8i>, cpol).
// ---------------------------------------------------------------------------
__device__ __forceinline__ void tdm_gather16(unsigned long long gaddr,
                                             unsigned lds_off,
                                             const int* __restrict__ xp, int t) {
  v4u g0;
  g0[0] = 0x80000001u;                                  // count=1, gather_mode=1, idx16
  g0[1] = lds_off;                                      // lds_addr (bytes)
  g0[2] = (unsigned)(gaddr & 0xFFFFFFFFull);            // global_addr lo
  g0[3] = (unsigned)((gaddr >> 32) & 0x01FFFFFFull)     // global_addr hi (57-bit)
        | 0x80000000u;                                  // type = 2 ("image")
  v8i g1;
  g1[0] = 0x00010000;                                   // wg_mask=0, data_size=2B
  g1[1] = (int)(((unsigned)EDIM) << 16);                // tensor_dim0 = 128
  g1[2] = (int)(((unsigned)VSZ & 0xFFFFu) << 16);       // tensor_dim1 = 50000 (lo16)
  g1[3] = (int)(((unsigned)EDIM) << 16);                // tensor_dim1 hi=0, tile_dim0=128
  g1[4] = 16;                                           // tile_dim1 = 16 valid indices
  g1[5] = EDIM;                                         // tensor_dim0_stride = 128
  g1[6] = 0;
  g1[7] = 0;
  v4i g2, g3;
  #pragma unroll
  for (int p = 0; p < 4; ++p) {
    unsigned a = (unsigned)xp[(2 * p) * TT + t];
    unsigned b = (unsigned)xp[(2 * p + 1) * TT + t];
    g2[p] = (int)((a & 0xFFFFu) | (b << 16));
  }
  #pragma unroll
  for (int p = 0; p < 4; ++p) {
    unsigned a = (unsigned)xp[(8 + 2 * p) * TT + t];
    unsigned b = (unsigned)xp[(8 + 2 * p + 1) * TT + t];
    g3[p] = (int)((a & 0xFFFFu) | (b << 16));
  }
  v8i g4;                                               // unused trailing operand
  #pragma unroll
  for (int p = 0; p < 8; ++p) g4[p] = 0;
  __builtin_amdgcn_tensor_load_to_lds(g0, g1, g2, g3, g4, 0);
}
#endif

// ---------------------------------------------------------------------------
// emb (f32) -> bf16 table in workspace (one-time, bandwidth-trivial)
// ---------------------------------------------------------------------------
__global__ void emb_to_bf16_kernel(const float* __restrict__ emb,
                                   unsigned short* __restrict__ out, int n) {
  int i = blockIdx.x * blockDim.x + threadIdx.x;
  if (i < n) out[i] = f32_to_bf16(emb[i]);
}

// ---------------------------------------------------------------------------
// Persistent BiLSTM kernel. Grid = 16 blocks x 256 threads.
//   blockIdx.x [0..7]  : forward LSTM, batch rows 16*blk .. +15
//   blockIdx.x [8..15] : backward LSTM
// Weights live in LDS as bf16 (2 x 128 KB); per-step gates via
// v_wmma_f32_16x16x32_bf16. Embedding tile for step t+1 is fetched by the
// Tensor Data Mover (gather mode) into a double-buffered LDS tile while
// step t computes -> 2 barriers per step, gather latency off critical path.
// ---------------------------------------------------------------------------
__global__ void lstm_dir_kernel(const int* __restrict__ x,
                                const float* __restrict__ emb,
                                const unsigned short* __restrict__ emb16,
                                const float* __restrict__ w_ih_f, const float* __restrict__ w_hh_f,
                                const float* __restrict__ b_ih_f, const float* __restrict__ b_hh_f,
                                const float* __restrict__ w_ih_b, const float* __restrict__ w_hh_b,
                                const float* __restrict__ b_ih_b, const float* __restrict__ b_hh_b,
                                float* __restrict__ h_all /* [T][B][H] */) {
  extern __shared__ char smem[];
  unsigned short* w_ih_s = (unsigned short*)(smem + OFF_WIH);
  unsigned short* w_hh_s = (unsigned short*)(smem + OFF_WHH);
  unsigned short* h_s    = (unsigned short*)(smem + OFF_H);
  float*          c_s    = (float*)(smem + OFF_C);
  float*          g_s    = (float*)(smem + OFF_G);
  float*          bias_s = (float*)(smem + OFF_BIAS);

  const int tid = threadIdx.x;
  const int dir = (int)(blockIdx.x >> 3);
  const int b0  = (int)(blockIdx.x & 7) * 16;

  const float* w_ih = dir ? w_ih_b : w_ih_f;
  const float* w_hh = dir ? w_hh_b : w_hh_f;
  const float* bi   = dir ? b_ih_b : b_ih_f;
  const float* bh   = dir ? b_hh_b : b_hh_f;
  const int*   xp   = x + (size_t)b0 * TT;

  // Stage weights (f32 -> bf16) and bias into LDS; zero h/c state.
  for (int i = tid; i < G4 * HD; i += 256) {
    w_ih_s[i] = f32_to_bf16(w_ih[i]);
    w_hh_s[i] = f32_to_bf16(w_hh[i]);
  }
  for (int i = tid; i < G4; i += 256) bias_s[i] = bi[i] + bh[i];
  for (int i = tid; i < 16 * HD; i += 256) { h_s[i] = 0; c_s[i] = 0.0f; }

  const int wv   = tid >> 5;        // wave 0..7 -> gate cols [64*wv, 64*wv+64)
  const int lane = tid & 31;
  const int lr   = lane & 15;
  const int half = lane >> 4;

#if USE_TDM
  const unsigned long long embaddr = (unsigned long long)emb16;
  if (wv == 0) {                    // prefill e-tile for the first step
    tdm_gather16(embaddr, OFF_E0, xp, dir ? TT - 1 : 0);
    __builtin_amdgcn_s_wait_tensorcnt(0);
  }
#endif
  __syncthreads();

  for (int ti = 0; ti < TT; ++ti) {
    const int t   = dir ? (TT - 1 - ti) : ti;
    const int cur = ti & 1;
    const unsigned short* e_sbuf =
        (const unsigned short*)(smem + OFF_E0 + (unsigned)cur * 4096u);

#if USE_TDM
    // Kick off the async gather for step t+1 into the other buffer (wave 0).
    if (wv == 0 && ti + 1 < TT) {
      tdm_gather16(embaddr, OFF_E0 + ((unsigned)(cur ^ 1)) * 4096u,
                   xp, dir ? (t - 1) : (t + 1));
    }
#else
    // Fallback: synchronous gather (f32 table), single buffer semantics.
    #pragma unroll
    for (int q = 0; q < 8; ++q) {
      int idx = tid + (q << 8);
      int m = idx >> 7, col = idx & 127;
      int tok = xp[m * TT + t];
      ((unsigned short*)(smem + OFF_E0 + (unsigned)cur * 4096u))[idx] =
          f32_to_bf16(emb[(size_t)tok * EDIM + col]);
    }
    __syncthreads();
#endif

    // ---- gates = e @ w_ih^T + h @ w_hh^T + bias  (WMMA bf16, f32 acc) --
    v8f acc[4];
    #pragma unroll
    for (int nt = 0; nt < 4; ++nt) {
      float bv = bias_s[(wv * 4 + nt) * 16 + lr];
      #pragma unroll
      for (int r = 0; r < 8; ++r) acc[nt][r] = bv;
    }
    #pragma unroll
    for (int kc = 0; kc < 4; ++kc) {
      const int k0 = kc << 5;
      // A operands (16x32 bf16): row M=lr; K = 16*half + e
      v16s ae, ah;
      #pragma unroll
      for (int e = 0; e < 16; ++e) {
        int kk = k0 + (half << 4) + e;
        ae[e] = (short)e_sbuf[lr * EDIM + kk];
        ah[e] = (short)h_s[lr * HD + kk];
      }
      const v16bf a_e = __builtin_bit_cast(v16bf, ae);
      const v16bf a_h = __builtin_bit_cast(v16bf, ah);
      #pragma unroll
      for (int nt = 0; nt < 4; ++nt) {
        const int g0i = (wv * 4 + nt) * 16;
        // B operand (32x16 bf16): N=lr, K = k0 + 16*half + e -> contiguous 32B
        const v16s bih = *(const v16s*)(w_ih_s + (size_t)(g0i + lr) * HD + k0 + (half << 4));
        const v16s bhh = *(const v16s*)(w_hh_s + (size_t)(g0i + lr) * HD + k0 + (half << 4));
        acc[nt] = __builtin_amdgcn_wmma_f32_16x16x32_bf16(
            false, a_e, false, __builtin_bit_cast(v16bf, bih), (short)0, acc[nt], false, false);
        acc[nt] = __builtin_amdgcn_wmma_f32_16x16x32_bf16(
            false, a_h, false, __builtin_bit_cast(v16bf, bhh), (short)0, acc[nt], false, false);
      }
    }
    // C/D layout: VGPR r -> M = r + 8*half, N = lr
    #pragma unroll
    for (int nt = 0; nt < 4; ++nt) {
      #pragma unroll
      for (int r = 0; r < 8; ++r)
        g_s[(r + (half << 3)) * G4 + (wv * 4 + nt) * 16 + lr] = acc[nt][r];
    }
    __syncthreads();

    // ---- activations + state update (gate order i,f,g,o) ---------------
    #pragma unroll
    for (int q = 0; q < 8; ++q) {
      int idx = tid + (q << 8);               // == m*128 + j
      int m = idx >> 7, j = idx & 127;
      float gi = g_s[m * G4 + j];
      float gf = g_s[m * G4 + 128 + j];
      float gg = g_s[m * G4 + 256 + j];
      float go = g_s[m * G4 + 384 + j];
      gi = 1.0f / (1.0f + __expf(-gi));
      gf = 1.0f / (1.0f + __expf(-gf));
      go = 1.0f / (1.0f + __expf(-go));
      gg = tanhf(gg);
      float c = gf * c_s[idx] + gi * gg;
      c_s[idx] = c;
      float h = go * tanhf(c);
      h_s[idx] = f32_to_bf16(h);
      h_all[((size_t)t * BB + (b0 + m)) * HDIM + dir * HD + j] = h;
    }
#if USE_TDM
    if (wv == 0) __builtin_amdgcn_s_wait_tensorcnt(0);  // e-tile for t+1 landed
#endif
    __syncthreads();
  }
}

// ---------------------------------------------------------------------------
// emissions[t,b,:] = h_all[t,b,:] @ W_tag^T + b_tag   (K=9, bandwidth-trivial)
// ---------------------------------------------------------------------------
__global__ void emissions_kernel(const float* __restrict__ h_all,
                                 const float* __restrict__ W_tag,
                                 const float* __restrict__ b_tag,
                                 float* __restrict__ emis) {
  __shared__ float Ws[KTAGS * HDIM];
  __shared__ float bs[KTAGS];
  for (int i = threadIdx.x; i < KTAGS * HDIM; i += blockDim.x) Ws[i] = W_tag[i];
  if (threadIdx.x < KTAGS) bs[threadIdx.x] = b_tag[threadIdx.x];
  __syncthreads();
  int idx = blockIdx.x * blockDim.x + threadIdx.x;   // (t*B + b)
  if (idx >= TT * BB) return;
  const float* hr = h_all + (size_t)idx * HDIM;
  float acc[KTAGS];
  #pragma unroll
  for (int k = 0; k < KTAGS; ++k) acc[k] = bs[k];
  for (int c = 0; c < HDIM; c += 4) {
    float4 hv = *(const float4*)(hr + c);
    #pragma unroll
    for (int k = 0; k < KTAGS; ++k) {
      acc[k] += hv.x * Ws[k * HDIM + c]     + hv.y * Ws[k * HDIM + c + 1]
              + hv.z * Ws[k * HDIM + c + 2] + hv.w * Ws[k * HDIM + c + 3];
    }
  }
  #pragma unroll
  for (int k = 0; k < KTAGS; ++k) emis[(size_t)idx * KTAGS + k] = acc[k];
}

// ---------------------------------------------------------------------------
// Viterbi decode (torchcrf, no mask). One thread per batch element.
// ---------------------------------------------------------------------------
__global__ void viterbi_kernel(const float* __restrict__ emis,
                               const float* __restrict__ start_trans,
                               const float* __restrict__ end_trans,
                               const float* __restrict__ trans,
                               int* __restrict__ bp,
                               float* __restrict__ out) {
  __shared__ float tr[KTAGS * KTAGS];
  __shared__ float st[KTAGS], et[KTAGS];
  if (threadIdx.x < KTAGS * KTAGS) tr[threadIdx.x] = trans[threadIdx.x];
  if (threadIdx.x < KTAGS) {
    st[threadIdx.x] = start_trans[threadIdx.x];
    et[threadIdx.x] = end_trans[threadIdx.x];
  }
  __syncthreads();
  int b = blockIdx.x * blockDim.x + threadIdx.x;
  if (b >= BB) return;
  float score[KTAGS], ns[KTAGS];
  #pragma unroll
  for (int j = 0; j < KTAGS; ++j) score[j] = st[j] + emis[(size_t)b * KTAGS + j];
  for (int t = 1; t < TT; ++t) {
    const float* em = emis + ((size_t)t * BB + b) * KTAGS;
    int* bpt = bp + ((size_t)(t - 1) * BB + b) * KTAGS;
    #pragma unroll
    for (int j = 0; j < KTAGS; ++j) {
      float best = score[0] + tr[j];
      int bi = 0;
      #pragma unroll
      for (int i = 1; i < KTAGS; ++i) {
        float cnd = score[i] + tr[i * KTAGS + j];
        if (cnd > best) { best = cnd; bi = i; }   // strict '>' == first argmax
      }
      ns[j] = best + em[j];
      bpt[j] = bi;
    }
    #pragma unroll
    for (int j = 0; j < KTAGS; ++j) score[j] = ns[j];
  }
  int last = 0;
  float bestf = score[0] + et[0];
  #pragma unroll
  for (int j = 1; j < KTAGS; ++j) {
    float cnd = score[j] + et[j];
    if (cnd > bestf) { bestf = cnd; last = j; }
  }
  out[(size_t)b * TT + (TT - 1)] = (float)last;
  for (int t = TT - 2; t >= 0; --t) {
    last = bp[((size_t)t * BB + b) * KTAGS + last];
    out[(size_t)b * TT + t] = (float)last;
  }
}

extern "C" void kernel_launch(void* const* d_in, const int* in_sizes, int n_in,
                              void* d_out, int out_size, void* d_ws, size_t ws_size,
                              hipStream_t stream) {
  const int*   x     = (const int*)  d_in[0];
  const float* emb   = (const float*)d_in[1];
  const float* wihf  = (const float*)d_in[2];
  const float* whhf  = (const float*)d_in[3];
  const float* bihf  = (const float*)d_in[4];
  const float* bhhf  = (const float*)d_in[5];
  const float* wihb  = (const float*)d_in[6];
  const float* whhb  = (const float*)d_in[7];
  const float* bihb  = (const float*)d_in[8];
  const float* bhhb  = (const float*)d_in[9];
  const float* Wtag  = (const float*)d_in[10];
  const float* btag  = (const float*)d_in[11];
  const float* strt  = (const float*)d_in[12];
  const float* endt  = (const float*)d_in[13];
  const float* trans = (const float*)d_in[14];

  // Workspace layout (~85 MB): emb_bf16 | h_all | emissions | backpointers
  char* ws = (char*)d_ws;
  const size_t emb16_bytes = (size_t)VSZ * EDIM * 2;                 // 12.8 MB
  const size_t hall_bytes  = (size_t)TT * BB * HDIM * 4;             // 67.1 MB
  const size_t emis_bytes  = (size_t)TT * BB * KTAGS * 4;            // 2.36 MB
  unsigned short* emb16 = (unsigned short*)ws;
  float* h_all = (float*)(ws + emb16_bytes);
  float* emis  = (float*)(ws + emb16_bytes + hall_bytes);
  int*   bp    = (int*)(ws + emb16_bytes + hall_bytes + emis_bytes);

  (void)hipFuncSetAttribute((const void*)lstm_dir_kernel,
                            hipFuncAttributeMaxDynamicSharedMemorySize, SMEM_BYTES);

  int n_emb = VSZ * EDIM;
  emb_to_bf16_kernel<<<(n_emb + 255) / 256, 256, 0, stream>>>(emb, emb16, n_emb);
  lstm_dir_kernel<<<16, 256, SMEM_BYTES, stream>>>(x, emb, emb16,
                                                   wihf, whhf, bihf, bhhf,
                                                   wihb, whhb, bihb, bhhb, h_all);
  emissions_kernel<<<(TT * BB + 255) / 256, 256, 0, stream>>>(h_all, Wtag, btag, emis);
  viterbi_kernel<<<1, BB, 0, stream>>>(emis, strt, endt, trans, bp, (float*)d_out);
}
